// EnergyTransformer_68736656605248
// MI455X (gfx1250) — matrix operand order repaired and checked by reading, not compiled, not run on hardware
//
#include <hip/hip_runtime.h>

#define STEPS 12
#define ALPHA 0.125f
#define EPSLN 1e-5f

// B=4, N=512, D=1024, H=16, DH=64, M=4096  (ROWS = B*N = 2048)

typedef __attribute__((ext_vector_type(16))) _Float16 v16h;
typedef __attribute__((ext_vector_type(8)))  float    v8f;

// ---------------------------------------------------------------------------
// WMMA fragment loaders (CDNA5 v_wmma_f32_16x16x32_f16 layouts, wave32)
// A (16x32, MxK): lanes 0-15 -> M=lane, K split {0..7,16..23}; lanes 16-31 -> +8
// B (32x16, KxN): lane -> N = lane&15, K split {0..7,16..23}; lanes 16-31 -> +8  (column-major load,
//                 so B source is stored as [N][K] row-major => contiguous)
// D (16x16 f32):  VGPR v -> M = v + 8*(lane>>4), N = lane&15
// ---------------------------------------------------------------------------
__device__ __forceinline__ v16h load_a(const _Float16* __restrict__ A, int lda,
                                       int m0, int k0, int lane) {
  v16h a;
  const _Float16* p = A + (size_t)(m0 + (lane & 15)) * lda + k0 + ((lane >> 4) << 3);
#pragma unroll
  for (int e = 0; e < 8; ++e) a[e] = p[e];
#pragma unroll
  for (int e = 0; e < 8; ++e) a[e + 8] = p[e + 16];
  return a;
}

// A[m,k] = Asrc[k*lda + m]   (transposed source)
__device__ __forceinline__ v16h load_aT(const _Float16* __restrict__ A, int lda,
                                        int m0, int k0, int lane) {
  v16h a;
  const int m  = m0 + (lane & 15);
  const int kb = k0 + ((lane >> 4) << 3);
#pragma unroll
  for (int e = 0; e < 8; ++e) a[e] = A[(size_t)(kb + e) * lda + m];
#pragma unroll
  for (int e = 0; e < 8; ++e) a[e + 8] = A[(size_t)(kb + e + 16) * lda + m];
  return a;
}

// B operand supplied as Bt = B^T stored [N][K] row-major, stride ldbt
__device__ __forceinline__ v16h load_bT(const _Float16* Bt, int ldbt,
                                        int k0, int n0, int lane) {
  v16h b;
  const _Float16* p = Bt + (size_t)(n0 + (lane & 15)) * ldbt + k0 + ((lane >> 4) << 3);
#pragma unroll
  for (int e = 0; e < 8; ++e) b[e] = p[e];
#pragma unroll
  for (int e = 0; e < 8; ++e) b[e + 8] = p[e + 16];
  return b;
}

// ---------------------------------------------------------------------------
// Generic batched WMMA GEMM: C[z] = A[z] * B[z]
// grid = (N/64, M/256, Z); block = 256 (8 waves, 1 wave -> 32x64 tile = 8 WMMA)
// k-loop is register double-buffered: fragments for k+32 are fetched before the
// WMMAs for k issue, so VMEM overlaps the matrix pipe.
// z = zo*NB + zi; operand base = zo*O + zi*I (element offsets)
// EPI: 0 = store f16 C
//      1 = relu -> store f16 C
//      2 = Cf32 += alpha * acc
//      3 = store f16 C and C^T (Ct, stride ldct)
// ---------------------------------------------------------------------------
template <int EPI, bool TRANSA>
__global__ __launch_bounds__(256) void gemm_wmma(
    const _Float16* __restrict__ A, int lda, size_t aO, size_t aI,
    const _Float16* __restrict__ Bt, int ldbt, size_t bO, size_t bI,
    void* __restrict__ Cp, int ldc, size_t cO, size_t cI,
    _Float16* __restrict__ Ct, int ldct, size_t ctO, size_t ctI,
    int Kdim, int NB, float alpha) {
  const int wave = threadIdx.x >> 5;
  const int lane = threadIdx.x & 31;
  const int zo = blockIdx.z / NB, zi = blockIdx.z % NB;
  const _Float16* Az = A + (size_t)zo * aO + (size_t)zi * aI;
  const _Float16* Bz = Bt + (size_t)zo * bO + (size_t)zi * bI;
  const int m0 = (blockIdx.y * 8 + wave) << 5;  // 2 m-tiles per wave
  const int n0 = blockIdx.x << 6;               // 4 n-tiles per wave

  v8f c[2][4] = {};
  v16h a[2], b[4];
  // prologue: fragments for k0 = 0
#pragma unroll
  for (int t = 0; t < 2; ++t)
    a[t] = TRANSA ? load_aT(Az, lda, m0 + (t << 4), 0, lane)
                  : load_a(Az, lda, m0 + (t << 4), 0, lane);
#pragma unroll
  for (int j = 0; j < 4; ++j) b[j] = load_bT(Bz, ldbt, 0, n0 + (j << 4), lane);

  for (int k0 = 0; k0 < Kdim; k0 += 32) {
    v16h an[2], bn[4];
    const int kn = k0 + 32;
    if (kn < Kdim) {  // prefetch next k-slice before consuming current
#pragma unroll
      for (int t = 0; t < 2; ++t)
        an[t] = TRANSA ? load_aT(Az, lda, m0 + (t << 4), kn, lane)
                       : load_a(Az, lda, m0 + (t << 4), kn, lane);
#pragma unroll
      for (int j = 0; j < 4; ++j) bn[j] = load_bT(Bz, ldbt, kn, n0 + (j << 4), lane);
    }
#pragma unroll
    for (int j = 0; j < 4; ++j) {
      c[0][j] = __builtin_amdgcn_wmma_f32_16x16x32_f16(false, a[0], false, b[j],
                                                       (short)0, c[0][j], false, false);
      c[1][j] = __builtin_amdgcn_wmma_f32_16x16x32_f16(false, a[1], false, b[j],
                                                       (short)0, c[1][j], false, false);
    }
    if (kn < Kdim) {
#pragma unroll
      for (int t = 0; t < 2; ++t) a[t] = an[t];
#pragma unroll
      for (int j = 0; j < 4; ++j) b[j] = bn[j];
    }
  }

  const int nn0 = n0 + (lane & 15);
#pragma unroll
  for (int t = 0; t < 2; ++t) {
    const int mr = m0 + (t << 4) + ((lane >> 4) << 3);
    if constexpr (EPI == 2) {
      float* Cz = (float*)Cp + (size_t)zo * cO + (size_t)zi * cI;
#pragma unroll
      for (int j = 0; j < 4; ++j)
#pragma unroll
        for (int v = 0; v < 8; ++v)
          Cz[(size_t)(mr + v) * ldc + nn0 + (j << 4)] += alpha * c[t][j][v];
    } else {
      _Float16* Cz = (_Float16*)Cp + (size_t)zo * cO + (size_t)zi * cI;
#pragma unroll
      for (int j = 0; j < 4; ++j) {
#pragma unroll
        for (int v = 0; v < 8; ++v) {
          float f = c[t][j][v];
          if constexpr (EPI == 1) f = f > 0.f ? f : 0.f;
          _Float16 hv = (_Float16)f;
          Cz[(size_t)(mr + v) * ldc + nn0 + (j << 4)] = hv;
          if constexpr (EPI == 3)
            Ct[(size_t)zo * ctO + (size_t)zi * ctI +
               (size_t)(nn0 + (j << 4)) * ldct + (mr + v)] = hv;
        }
      }
    }
  }
}

// ---------------------------------------------------------------------------
// Fused attention scores + softmax: per (b,h), per 16-query strip.
// S[i, j] = (1/8) * sum_e K[i,e] * Q[j,e]  via WMMA into LDS (512 x 16 f32),
// softmax over keys i per column j, writes P^T[j, i] in f16.
// The 16x64 Q strip (2 KB) is shared by all 8 waves -> staged once into LDS
// with CDNA5 async global->LDS copies (ASYNCcnt), B fragments then come from
// LDS. K-row A fragments are software-pipelined across the 4 i-tiles.
// grid = (N/16 = 32, B*H = 64); block = 256 (8 waves x 4 i-tiles)
// ---------------------------------------------------------------------------
__global__ __launch_bounds__(256) void attn_softmax(
    const _Float16* __restrict__ Kh, const _Float16* __restrict__ Qh,
    _Float16* __restrict__ PT) {
  __shared__ float Sld[512 * 16];
  __shared__ float red[256];
  __shared__ __align__(16) _Float16 Qs[16 * 64];  // staged Q strip [j][e]
  const int bh = blockIdx.y;
  const int j0 = blockIdx.x << 4;
  const _Float16* Km = Kh + (size_t)bh * 512 * 64;
  const _Float16* Qm = Qh + (size_t)bh * 512 * 64;
  const int wave = threadIdx.x >> 5;
  const int lane = threadIdx.x & 31;

  // Stage Q strip into LDS: 2048 bytes = 128 x 16B async chunks.
  if (threadIdx.x < 128) {
    const unsigned long long gaddr =
        (unsigned long long)(uintptr_t)(Qm + (size_t)j0 * 64) + threadIdx.x * 16u;
    const unsigned int laddr =
        (unsigned int)(uintptr_t)(&Qs[0]) + threadIdx.x * 16u;
    asm volatile("global_load_async_to_lds_b128 %0, %1, off"
                 :: "v"(laddr), "v"(gaddr) : "memory");
  }
  asm volatile("s_wait_asynccnt 0" ::: "memory");
  __syncthreads();

  // Phase 1: WMMA score strip -> LDS (A loads pipelined across i-tiles,
  // B fragments are loop-invariant and read from the staged LDS strip)
  const v16h b0 = load_bT(&Qs[0], 64, 0, 0, lane);
  const v16h b1 = load_bT(&Qs[0], 64, 32, 0, lane);
  v16h a0 = load_a(Km, 64, (wave << 6), 0, lane);
  v16h a1 = load_a(Km, 64, (wave << 6), 32, lane);
#pragma unroll
  for (int t = 0; t < 4; ++t) {
    const int i0 = (wave << 6) + (t << 4);
    v16h a0n, a1n;
    if (t < 3) {  // prefetch next tile's K rows
      a0n = load_a(Km, 64, i0 + 16, 0, lane);
      a1n = load_a(Km, 64, i0 + 16, 32, lane);
    }
    v8f acc = {};
    acc = __builtin_amdgcn_wmma_f32_16x16x32_f16(false, a0, false, b0, (short)0,
                                                 acc, false, false);
    acc = __builtin_amdgcn_wmma_f32_16x16x32_f16(false, a1, false, b1, (short)0,
                                                 acc, false, false);
    const int nn = lane & 15;
    const int mr = i0 + ((lane >> 4) << 3);
#pragma unroll
    for (int v = 0; v < 8; ++v) Sld[(mr + v) * 16 + nn] = acc[v] * 0.125f;
    if (t < 3) { a0 = a0n; a1 = a1n; }
  }
  __syncthreads();

  // Phase 2: softmax over i (512) per column j (16)
  const int j = threadIdx.x & 15;
  const int r = threadIdx.x >> 4;  // 0..15
  float mx = -1e30f;
  for (int i = r; i < 512; i += 16) mx = fmaxf(mx, Sld[i * 16 + j]);
  red[r * 16 + j] = mx;
  __syncthreads();
  for (int s = 8; s > 0; s >>= 1) {
    if (r < s) red[r * 16 + j] = fmaxf(red[r * 16 + j], red[(r + s) * 16 + j]);
    __syncthreads();
  }
  mx = red[j];
  __syncthreads();
  float sum = 0.f;
  for (int i = r; i < 512; i += 16) {
    float e = __expf(Sld[i * 16 + j] - mx);
    Sld[i * 16 + j] = e;
    sum += e;
  }
  red[r * 16 + j] = sum;
  __syncthreads();
  for (int s = 8; s > 0; s >>= 1) {
    if (r < s) red[r * 16 + j] += red[(r + s) * 16 + j];
    __syncthreads();
  }
  const float inv = 1.0f / red[j];
  _Float16* PTr = PT + (size_t)bh * 512 * 512 + (size_t)(j0 + j) * 512;
  for (int i = r; i < 512; i += 16) PTr[i] = (_Float16)(Sld[i * 16 + j] * inv);
}

// ---------------------------------------------------------------------------
// LayerNorm over D=1024, one block per row, 256 threads; f16 output.
// ---------------------------------------------------------------------------
__global__ __launch_bounds__(256) void layernorm_k(
    const float* __restrict__ X, const float* __restrict__ gamma,
    const float* __restrict__ beta, _Float16* __restrict__ G) {
  __shared__ float s1[256], s2[256];
  const int row = blockIdx.x;
  const int tid = threadIdx.x;
  const float* x = X + (size_t)row * 1024;
  float v[4], sum = 0.f;
#pragma unroll
  for (int q = 0; q < 4; ++q) { v[q] = x[tid + (q << 8)]; sum += v[q]; }
  s1[tid] = sum;
  __syncthreads();
  for (int s = 128; s > 0; s >>= 1) {
    if (tid < s) s1[tid] += s1[tid + s];
    __syncthreads();
  }
  const float mean = s1[0] * (1.0f / 1024.0f);
  float sq = 0.f;
#pragma unroll
  for (int q = 0; q < 4; ++q) { float d = v[q] - mean; sq += d * d; }
  s2[tid] = sq;
  __syncthreads();
  for (int s = 128; s > 0; s >>= 1) {
    if (tid < s) s2[tid] += s2[tid + s];
    __syncthreads();
  }
  const float rstd = rsqrtf(s2[0] * (1.0f / 1024.0f) + EPSLN);
  _Float16* g = G + (size_t)row * 1024;
#pragma unroll
  for (int q = 0; q < 4; ++q) {
    const int d = tid + (q << 8);
    g[d] = (_Float16)(gamma[d] * (v[q] - mean) * rstd + beta[d]);
  }
}

// ---------------------------------------------------------------------------
// One-time conversions
// ---------------------------------------------------------------------------
__global__ void cvt_w(const float* __restrict__ wq, const float* __restrict__ wk,
                      _Float16* __restrict__ wqT, _Float16* __restrict__ wkT,
                      _Float16* __restrict__ wqC, _Float16* __restrict__ wkC) {
  const int idx = blockIdx.x * 256 + threadIdx.x;  // H*D*DH = 1048576
  if (idx >= 16 * 1024 * 64) return;
  const int e = idx & 63, d = (idx >> 6) & 1023, h = idx >> 16;
  const float q = wq[idx], k = wk[idx];
  const size_t t = ((size_t)((h << 6) + e)) * 1024 + d;   // [H*DH, D]
  const size_t c = (size_t)d * 1024 + (h << 6) + e;       // [D, H*DH]
  wqT[t] = (_Float16)q; wkT[t] = (_Float16)k;
  wqC[c] = (_Float16)q; wkC[c] = (_Float16)k;
}

__global__ void cvt_xi(const float* __restrict__ xi, _Float16* __restrict__ xi_h,
                       _Float16* __restrict__ xiT) {
  const int idx = blockIdx.x * 256 + threadIdx.x;  // M*D = 4194304
  if (idx >= 4096 * 1024) return;
  const int d = idx & 1023, m = idx >> 10;
  const float v = xi[idx];
  xi_h[idx] = (_Float16)v;                 // [M, D]
  xiT[(size_t)d * 4096 + m] = (_Float16)v; // [D, M]
}

__global__ void copy_x(const float* __restrict__ src, float* __restrict__ dst) {
  const int i = blockIdx.x * 256 + threadIdx.x;
  if (i < 2048 * 1024) dst[i] = src[i];
}

__global__ void final_add(const float* __restrict__ xc, const float* __restrict__ x0,
                          const float* __restrict__ ss, float* __restrict__ out) {
  const int i = blockIdx.x * 256 + threadIdx.x;
  if (i < 2048 * 1024) out[i] = xc[i] + ss[0] * x0[i];
}

// ---------------------------------------------------------------------------
extern "C" void kernel_launch(void* const* d_in, const int* in_sizes, int n_in,
                              void* d_out, int out_size, void* d_ws, size_t ws_size,
                              hipStream_t stream) {
  (void)in_sizes; (void)n_in; (void)out_size; (void)ws_size;
  const float* x_in  = (const float*)d_in[0];
  const float* gamma = (const float*)d_in[1];
  const float* beta  = (const float*)d_in[2];
  const float* wq    = (const float*)d_in[3];
  const float* wk    = (const float*)d_in[4];
  const float* xi    = (const float*)d_in[5];
  const float* ss    = (const float*)d_in[6];

  char* w = (char*)d_ws;
  size_t off = 0;
  auto take = [&](size_t bytes) {
    void* p = w + off;
    off = (off + bytes + 255) & ~(size_t)255;
    return p;
  };
  float*    x_cur = (float*)   take(2048ull * 1024 * 4);   // 8 MB
  _Float16* g_h   = (_Float16*)take(2048ull * 1024 * 2);   // 4 MB
  _Float16* wqT_h = (_Float16*)take(1024ull * 1024 * 2);   // [H*DH, D]
  _Float16* wkT_h = (_Float16*)take(1024ull * 1024 * 2);
  _Float16* wqC   = (_Float16*)take(1024ull * 1024 * 2);   // [D, H*DH]
  _Float16* wkC   = (_Float16*)take(1024ull * 1024 * 2);
  _Float16* xi_h  = (_Float16*)take(4096ull * 1024 * 2);   // [M, D]
  _Float16* xiT_h = (_Float16*)take(4096ull * 1024 * 2);   // [D, M]
  _Float16* Qh    = (_Float16*)take(64ull * 512 * 64 * 2); // [BH, N, DH]
  _Float16* Kh    = (_Float16*)take(64ull * 512 * 64 * 2);
  _Float16* QhT   = (_Float16*)take(64ull * 64 * 512 * 2); // [BH, DH, N]
  _Float16* KhT   = (_Float16*)take(64ull * 64 * 512 * 2);
  _Float16* PT    = (_Float16*)take(64ull * 512 * 512 * 2); // 32 MB
  _Float16* dq    = (_Float16*)take(2048ull * 1024 * 2);   // [B, N, H*DH]
  _Float16* dk    = (_Float16*)take(2048ull * 1024 * 2);
  _Float16* Hm    = (_Float16*)take(2048ull * 4096 * 2);   // 16 MB

  cvt_w  <<<4096,  256, 0, stream>>>(wq, wk, wqT_h, wkT_h, wqC, wkC);
  cvt_xi <<<16384, 256, 0, stream>>>(xi, xi_h, xiT_h);
  copy_x <<<8192,  256, 0, stream>>>(x_in, x_cur);

  for (int s = 0; s < STEPS; ++s) {
    layernorm_k<<<2048, 256, 0, stream>>>(x_cur, gamma, beta, g_h);
    // Q = g * Wq, K = g * Wk   (M=512,N=64,K=1024 per (b,h)); dual-store C,C^T
    gemm_wmma<3, false><<<dim3(1, 2, 64), 256, 0, stream>>>(
        g_h, 1024, 524288, 0, wqT_h, 1024, 0, 65536,
        Qh, 64, 524288, 32768, QhT, 512, 524288, 32768, 1024, 16, 0.f);
    gemm_wmma<3, false><<<dim3(1, 2, 64), 256, 0, stream>>>(
        g_h, 1024, 524288, 0, wkT_h, 1024, 0, 65536,
        Kh, 64, 524288, 32768, KhT, 512, 524288, 32768, 1024, 16, 0.f);
    // scores + softmax -> P^T
    attn_softmax<<<dim3(32, 64), 256, 0, stream>>>(Kh, Qh, PT);
    // dq = P^T * K  (M=512,N=64,K=512); out packed [B,N,H*DH]
    gemm_wmma<0, false><<<dim3(1, 2, 64), 256, 0, stream>>>(
        PT, 512, 4194304, 262144, KhT, 512, 524288, 32768,
        dq, 1024, 524288, 64, nullptr, 0, 0, 0, 512, 16, 0.f);
    // dk = P * Q = (P^T)^T * Q
    gemm_wmma<0, true><<<dim3(1, 2, 64), 256, 0, stream>>>(
        PT, 512, 4194304, 262144, QhT, 512, 524288, 32768,
        dk, 1024, 524288, 64, nullptr, 0, 0, 0, 512, 16, 0.f);
    // x += ALPHA * dq * WqT_cat   (M=512,N=1024,K=1024 per b)
    gemm_wmma<2, false><<<dim3(16, 2, 4), 256, 0, stream>>>(
        dq, 1024, 524288, 0, wqC, 1024, 0, 0,
        x_cur, 1024, 524288, 0, nullptr, 0, 0, 0, 1024, 1, ALPHA);
    gemm_wmma<2, false><<<dim3(16, 2, 4), 256, 0, stream>>>(
        dk, 1024, 524288, 0, wkC, 1024, 0, 0,
        x_cur, 1024, 524288, 0, nullptr, 0, 0, 0, 1024, 1, ALPHA);
    layernorm_k<<<2048, 256, 0, stream>>>(x_cur, gamma, beta, g_h);
    // Hm = relu(g * xi^T)  (M=2048,N=4096,K=1024)
    gemm_wmma<1, false><<<dim3(64, 8, 1), 256, 0, stream>>>(
        g_h, 1024, 0, 0, xi_h, 1024, 0, 0,
        Hm, 4096, 0, 0, nullptr, 0, 0, 0, 1024, 1, 0.f);
    // x += ALPHA * Hm * xi  (M=2048,N=1024,K=4096)
    gemm_wmma<2, false><<<dim3(16, 8, 1), 256, 0, stream>>>(
        Hm, 4096, 0, 0, xiT_h, 4096, 0, 0,
        x_cur, 1024, 0, 0, nullptr, 0, 0, 0, 4096, 1, ALPHA);
  }

  final_add<<<8192, 256, 0, stream>>>(x_cur, x_in, ss, (float*)d_out);
}
